// DGCNN_45174466019475
// MI455X (gfx1250) — matrix operand order repaired and checked by reading, not compiled, not run on hardware
//
#include <hip/hip_runtime.h>
#include <hip/hip_bf16.h>

typedef __attribute__((ext_vector_type(16))) _Float16 v16h;
typedef __attribute__((ext_vector_type(8)))  _Float16 v8h;
typedef __attribute__((ext_vector_type(8)))  float    v8f;

#define BB   4
#define NP   2048
#define KNN  20
#define CTOT 512
#define S_EDGE (BB*NP*KNN)   // 163840
#define S_PT   (BB*NP)       // 8192

// ---------- elementwise / prep kernels ----------

__global__ void k_sqnorm(const float* __restrict__ src, float* __restrict__ xx,
                         int C, int Cs, int coff) {
  int t = blockIdx.x*blockDim.x + threadIdx.x;
  if (t >= BB*NP) return;
  int b = t / NP, n = t % NP;
  float s = 0.f;
  for (int c = 0; c < C; ++c) {
    float v = src[((size_t)(b*Cs + coff + c))*NP + n];
    s += v*v;
  }
  xx[t] = s;
}

// feature -> xT[b][n][Cp] f16, K-contiguous, zero padded to Cp (multiple of 32)
__global__ void k_featT16(const float* __restrict__ src, _Float16* __restrict__ xT,
                          int C, int Cp, int Cs, int coff) {
  int t = blockIdx.x*blockDim.x + threadIdx.x;
  if (t >= BB*NP) return;
  int b = t / NP, n = t % NP;
  _Float16* o = xT + (size_t)t * Cp;
  for (int k = 0; k < Cp; ++k)
    o[k] = (k < C) ? (_Float16)src[((size_t)(b*Cs + coff + k))*NP + n] : (_Float16)0.f;
}

// weights (MxKd f32 row-major) -> (MxKp f16 row-major, zero padded)
__global__ void k_wpad16(const float* __restrict__ w, _Float16* __restrict__ wh,
                         int M, int Kd, int Kp) {
  int t = blockIdx.x*blockDim.x + threadIdx.x;
  if (t >= M*Kp) return;
  int k = t % Kp, m = t / Kp;
  wh[t] = (k < Kd) ? (_Float16)w[(size_t)m*Kd + k] : (_Float16)0.f;
}

// ---------- pairwise (-squared distance) via WMMA: one wave = 16x16 tile ----------
// A(m,k) = xT[n0+m][k]  (rows of the n-tile), B(k,n) = xT[m0+n][k]  (rows of m-tile)

__global__ void k_pairwise(const _Float16* __restrict__ xT, const float* __restrict__ xx,
                           float* __restrict__ pd, int Cp) {
  int n0 = blockIdx.x * 16;        // output rows (point n)
  int m0 = blockIdx.y * 16;        // output cols (point m)
  int b  = blockIdx.z;
  int lane = threadIdx.x;
  int g = lane >> 4, r = lane & 15;
  const _Float16* xb = xT + (size_t)b * NP * Cp;
  const _Float16* rowA = xb + (size_t)(n0 + r) * Cp;   // A fragment source row
  const _Float16* rowB = xb + (size_t)(m0 + r) * Cp;   // B fragment source row
  v8f acc = {};
  for (int k0 = 0; k0 < Cp; k0 += 32) {
    v8h alo = *(const v8h*)(rowA + k0 + 16*g);
    v8h ahi = *(const v8h*)(rowA + k0 + 16*g + 8);
    v16h a = __builtin_shufflevector(alo, ahi, 0,1,2,3,4,5,6,7,8,9,10,11,12,13,14,15);
    v16h bm = *(const v16h*)(rowB + k0 + 16*g);
    acc = __builtin_amdgcn_wmma_f32_16x16x32_f16(false, a, false, bm,
                                                 (short)0, acc, false, false);
  }
#pragma unroll
  for (int v = 0; v < 8; ++v) {
    int row = n0 + (g << 3) + v;
    int col = m0 + r;
    pd[((size_t)b*NP + row)*NP + col] = 2.f*acc[v] - xx[b*NP + row] - xx[b*NP + col];
  }
}

// ---------- top-k (k=20 largest of -dist, stable lowest-index-first ties) ----------

__global__ void k_topk(const float* __restrict__ pd, int* __restrict__ idx) {
  int t = blockIdx.x*blockDim.x + threadIdx.x;
  if (t >= BB*NP) return;
  const float* row = pd + (size_t)t * NP;
  float best[KNN]; int bi[KNN];
#pragma unroll
  for (int j = 0; j < KNN; ++j) { best[j] = -3.4e38f; bi[j] = 0; }
  for (int m = 0; m < NP; ++m) {
    float v = row[m];
    if (v > best[KNN-1]) {
      int j = KNN - 1;
      while (j > 0 && best[j-1] < v) { best[j] = best[j-1]; bi[j] = bi[j-1]; --j; }
      best[j] = v; bi[j] = m;
    }
  }
  int* o = idx + (size_t)t * KNN;
  for (int j = 0; j < KNN; ++j) o[j] = bi[j];
}

// ---------- edge features, transposed: E_T[s][Kp] = [diff(C) | center(C) | 0 pad] ----------

__global__ void k_edge(const float* __restrict__ src, const int* __restrict__ idx,
                       _Float16* __restrict__ ET, int C, int Kp, int Cs, int coff) {
  int s = blockIdx.x*blockDim.x + threadIdx.x;
  if (s >= S_EDGE) return;
  int n = (s / KNN) % NP; int b = s / (KNN * NP);
  int nb = idx[s];
  _Float16* o = ET + (size_t)s * Kp;
  for (int c = 0; c < C; ++c) {
    const float* xc = src + ((size_t)(b*Cs + coff + c))*NP;
    float vc = xc[n], vn = xc[nb];
    o[c]     = (_Float16)(vn - vc);
    o[C + c] = (_Float16)vc;
  }
  for (int c = 2*C; c < Kp; ++c) o[c] = (_Float16)0.f;
}

// ---------- WMMA GEMM: Y(MxS) = W(MxKp f16 row-major) * X_T(SxKp f16)^T ----------
// one wave computes a 64x16 block of Y (4 M-subtiles reuse the streamed B fragment)

__global__ void k_gemm(const _Float16* __restrict__ W, const _Float16* __restrict__ XT,
                       float* __restrict__ Y, int M, int Kp, int S) {
  int s0 = blockIdx.x * 16;
  int mb = blockIdx.y * 64;
  int lane = threadIdx.x;
  int g = lane >> 4, r = lane & 15;
  const _Float16* rowB = XT + (size_t)(s0 + r) * Kp;
  v8f acc[4] = {};
  for (int k0 = 0; k0 < Kp; k0 += 32) {
    __builtin_prefetch(rowB + k0 + 32, 0, 1);
    v16h bm = *(const v16h*)(rowB + k0 + 16*g);
#pragma unroll
    for (int i = 0; i < 4; ++i) {
      const _Float16* rowA = W + (size_t)(mb + i*16 + r) * Kp + k0;
      v8h alo = *(const v8h*)(rowA + 16*g);
      v8h ahi = *(const v8h*)(rowA + 16*g + 8);
      v16h a = __builtin_shufflevector(alo, ahi, 0,1,2,3,4,5,6,7,8,9,10,11,12,13,14,15);
      acc[i] = __builtin_amdgcn_wmma_f32_16x16x32_f16(false, a, false, bm,
                                                      (short)0, acc[i], false, false);
    }
  }
#pragma unroll
  for (int i = 0; i < 4; ++i)
#pragma unroll
    for (int v = 0; v < 8; ++v) {
      int m = mb + i*16 + (g << 3) + v;
      Y[(size_t)m*S + s0 + r] = acc[i][v];
    }
}

// ---------- per-channel batch stats (sum, sumsq) ----------

__global__ void k_stats(const float* __restrict__ Y, float* __restrict__ st,
                        int M, int S) {
  int m = blockIdx.x;
  __shared__ float s1[256], s2[256];
  float a = 0.f, b = 0.f;
  for (int i = threadIdx.x; i < S; i += 256) {
    float v = Y[(size_t)m*S + i]; a += v; b += v*v;
  }
  s1[threadIdx.x] = a; s2[threadIdx.x] = b; __syncthreads();
  for (int o = 128; o > 0; o >>= 1) {
    if (threadIdx.x < o) { s1[threadIdx.x] += s1[threadIdx.x+o]; s2[threadIdx.x] += s2[threadIdx.x+o]; }
    __syncthreads();
  }
  if (threadIdx.x == 0) { st[m] = s1[0]; st[M + m] = s2[0]; }
}

// ---------- BN + LeakyReLU + max over k  ->  F(B, 512, N) at channel offset ----------

__global__ void k_bnmax(const float* __restrict__ Y, const float* __restrict__ st,
                        const float* __restrict__ gw, const float* __restrict__ bw,
                        float* __restrict__ F, int M, int coff) {
  int t = blockIdx.x*blockDim.x + threadIdx.x;
  if (t >= BB*NP*M) return;
  int m = t % M; int n = (t / M) % NP; int b = t / (M * NP);
  float S = (float)S_EDGE;
  float mean = st[m] / S;
  float var  = st[M + m] / S - mean*mean;
  float inv  = rsqrtf(var + 1e-5f);
  float gm = gw[m] * inv, bt = bw[m] - mean * gm;
  const float* yr = Y + (size_t)m*S_EDGE + (size_t)(b*NP + n)*KNN;
  float mx = -3.4e38f;
  for (int j = 0; j < KNN; ++j) {
    float v = yr[j]*gm + bt;
    v = v > 0.f ? v : 0.2f*v;
    mx = fmaxf(mx, v);
  }
  F[((size_t)(b*CTOT + coff + m))*NP + n] = mx;
}

// ---------- F(B,512,N) -> FhT[(b*N+n)][512] f16 for conv5 ----------

__global__ void k_fT16(const float* __restrict__ F, _Float16* __restrict__ FhT) {
  int t = blockIdx.x*blockDim.x + threadIdx.x;
  if (t >= BB*NP) return;
  int b = t / NP, n = t % NP;
  _Float16* o = FhT + (size_t)t * CTOT;
  for (int c = 0; c < CTOT; ++c)
    o[c] = (_Float16)F[((size_t)(b*CTOT + c))*NP + n];
}

// ---------- BN5 + LeakyReLU + global max & mean pool -> z(B, 2048) ----------

__global__ void k_pool(const float* __restrict__ H, const float* __restrict__ st,
                       const float* __restrict__ gw, const float* __restrict__ bw,
                       float* __restrict__ z) {
  int b = blockIdx.x / 1024;
  int m = blockIdx.x % 1024;
  float S = (float)S_PT;
  float mean = st[m] / S;
  float var  = st[1024 + m] / S - mean*mean;
  float inv  = rsqrtf(var + 1e-5f);
  float gm = gw[m]*inv, bt = bw[m] - mean*gm;
  __shared__ float smax[256], ssum[256];
  float mx = -3.4e38f, sm = 0.f;
  for (int n = threadIdx.x; n < NP; n += 256) {
    float v = H[(size_t)m*S_PT + b*NP + n]*gm + bt;
    v = v > 0.f ? v : 0.2f*v;
    mx = fmaxf(mx, v); sm += v;
  }
  smax[threadIdx.x] = mx; ssum[threadIdx.x] = sm; __syncthreads();
  for (int o = 128; o > 0; o >>= 1) {
    if (threadIdx.x < o) {
      smax[threadIdx.x] = fmaxf(smax[threadIdx.x], smax[threadIdx.x+o]);
      ssum[threadIdx.x] += ssum[threadIdx.x+o];
    }
    __syncthreads();
  }
  if (threadIdx.x == 0) { z[b*2048 + m] = smax[0]; z[b*2048 + 1024 + m] = ssum[0]/(float)NP; }
}

// ---------- small dense layers (M=4 rows: scalar) ----------

__global__ void k_fc(const float* __restrict__ zin, const float* __restrict__ w,
                     const float* __restrict__ bias, float* __restrict__ zout,
                     int In, int Out) {
  int t = blockIdx.x*blockDim.x + threadIdx.x;
  if (t >= BB*Out) return;
  int o = t % Out, b = t / Out;
  float s = bias[o];
  for (int i = 0; i < In; ++i) s += zin[b*In + i] * w[(size_t)o*In + i];
  zout[t] = s;
}

// ---------- Student-t cluster assignment (ALPHA=1 -> 1/(1+d2), normalized) ----------

__global__ void k_cluster(const float* __restrict__ e, const float* __restrict__ cw,
                          float* __restrict__ q, float* __restrict__ xd) {
  int b = blockIdx.x;
  __shared__ float sh[256];
  float loc = 0.f;
  for (int j = threadIdx.x; j < 800; j += 256) {
    float s = 0.f;
    for (int c = 0; c < 10; ++c) { float d = e[b*10 + c] - cw[j*10 + c]; s += d*d; }
    if (xd) xd[b*800 + j] = s;
    float qr = 1.f / (1.f + s);
    q[b*800 + j] = qr;
    loc += qr;
  }
  sh[threadIdx.x] = loc; __syncthreads();
  for (int o = 128; o > 0; o >>= 1) {
    if (threadIdx.x < o) sh[threadIdx.x] += sh[threadIdx.x+o];
    __syncthreads();
  }
  float tot = sh[0];
  __syncthreads();
  for (int j = threadIdx.x; j < 800; j += 256) q[b*800 + j] /= tot;
}

__global__ void k_sim(const float* __restrict__ e1, const float* __restrict__ e2,
                      float* __restrict__ sim) {
  int b = threadIdx.x;
  if (b >= BB) return;
  float s = 0.f;
  for (int c = 0; c < 10; ++c) { float d = e1[b*10 + c] - e2[b*10 + c] + 1e-6f; s += d*d; }
  sim[b] = sqrtf(s);
}

// ---------- host orchestration ----------

extern "C" void kernel_launch(void* const* d_in, const int* in_sizes, int n_in,
                              void* d_out, int out_size, void* d_ws, size_t ws_size,
                              hipStream_t stream) {
  const float* xin[2] = {(const float*)d_in[0], (const float*)d_in[1]};
  const float* W[4]  = {(const float*)d_in[2], (const float*)d_in[5], (const float*)d_in[8],  (const float*)d_in[11]};
  const float* G[4]  = {(const float*)d_in[3], (const float*)d_in[6], (const float*)d_in[9],  (const float*)d_in[12]};
  const float* Bw[4] = {(const float*)d_in[4], (const float*)d_in[7], (const float*)d_in[10], (const float*)d_in[13]};
  const float* w5 = (const float*)d_in[14]; const float* g5 = (const float*)d_in[15];
  const float* b5 = (const float*)d_in[16];
  const float* lw1 = (const float*)d_in[17]; const float* lb1 = (const float*)d_in[18];
  const float* lw2 = (const float*)d_in[19]; const float* lb2 = (const float*)d_in[20];
  const float* lw3 = (const float*)d_in[21]; const float* lb3 = (const float*)d_in[22];
  const float* cw  = (const float*)d_in[23];

  float* out = (float*)d_out;
  float* o_sim  = out;                         // 4
  float* o_q[2] = {out + 4, out + 3204};       // 3200 each
  float* o_e[2] = {out + 6404, out + 6444};    // 40 each
  float* o_xd   = out + 6484;                  // 3200

  char* ws = (char*)d_ws;
  size_t off = 0;
  auto alloc = [&](size_t bytes) { size_t r = off; off += (bytes + 255) & ~(size_t)255; return r; };
  float*    xx  = (float*)   (ws + alloc(sizeof(float)    * (size_t)BB*NP));
  float*    pd  = (float*)   (ws + alloc(sizeof(float)    * (size_t)BB*NP*NP));
  int*      idx = (int*)     (ws + alloc(sizeof(int)      * (size_t)BB*NP*KNN));
  _Float16* xT  = (_Float16*)(ws + alloc(sizeof(_Float16) * (size_t)BB*NP*128));
  _Float16* ET  = (_Float16*)(ws + alloc(sizeof(_Float16) * (size_t)S_EDGE*512));
  float*    Y   = (float*)   (ws + alloc(sizeof(float)    * (size_t)256*S_EDGE));
  float*    F   = (float*)   (ws + alloc(sizeof(float)    * (size_t)BB*CTOT*NP));
  _Float16* FhT = (_Float16*)(ws + alloc(sizeof(_Float16) * (size_t)S_PT*CTOT));
  float*    H   = (float*)   (ws + alloc(sizeof(float)    * (size_t)1024*S_PT));
  float*    st  = (float*)   (ws + alloc(sizeof(float)    * 2*1024));
  _Float16* wh  = (_Float16*)(ws + alloc(sizeof(_Float16) * (size_t)1024*512));
  float*    z   = (float*)   (ws + alloc(sizeof(float)    * (size_t)BB*2048));
  float*    t1  = (float*)   (ws + alloc(sizeof(float)    * (size_t)BB*256));
  float*    t2  = (float*)   (ws + alloc(sizeof(float)    * (size_t)BB*64));

  const int Cin[4]     = {3, 64, 64, 128};
  const int Cp[4]      = {32, 64, 64, 128};        // kNN K padded to multiple of 32
  const int Cout[4]    = {64, 64, 128, 256};
  const int KpE[4]     = {32, 128, 256, 512};      // 2C padded to multiple of 32
  const int coffIn[4]  = {0, 0, 64, 128};
  const int coffOut[4] = {0, 64, 128, 256};

  for (int which = 0; which < 2; ++which) {
    const float* x = xin[which];
    for (int l = 0; l < 4; ++l) {
      const float* src = (l == 0) ? x : F;
      int Cs = (l == 0) ? 3 : CTOT;
      int C = Cin[l], co = coffIn[l];

      k_sqnorm<<<(BB*NP + 255)/256, 256, 0, stream>>>(src, xx, C, Cs, co);
      k_featT16<<<(BB*NP + 255)/256, 256, 0, stream>>>(src, xT, C, Cp[l], Cs, co);

      dim3 pg(NP/16, NP/16, BB);
      k_pairwise<<<pg, 32, 0, stream>>>(xT, xx, pd, Cp[l]);
      k_topk<<<(BB*NP + 255)/256, 256, 0, stream>>>(pd, idx);
      k_edge<<<(S_EDGE + 255)/256, 256, 0, stream>>>(src, idx, ET, C, KpE[l], Cs, co);

      int wn = Cout[l] * KpE[l];
      k_wpad16<<<(wn + 255)/256, 256, 0, stream>>>(W[l], wh, Cout[l], 2*C, KpE[l]);
      dim3 gg(S_EDGE/16, Cout[l]/64);
      k_gemm<<<gg, 32, 0, stream>>>(wh, ET, Y, Cout[l], KpE[l], S_EDGE);

      k_stats<<<Cout[l], 256, 0, stream>>>(Y, st, Cout[l], S_EDGE);
      int tm = BB*NP*Cout[l];
      k_bnmax<<<(tm + 255)/256, 256, 0, stream>>>(Y, st, G[l], Bw[l], F, Cout[l], coffOut[l]);
    }

    k_fT16<<<(BB*NP + 255)/256, 256, 0, stream>>>(F, FhT);
    k_wpad16<<<(1024*512 + 255)/256, 256, 0, stream>>>(w5, wh, 1024, 512, 512);
    dim3 g5g(S_PT/16, 1024/64);
    k_gemm<<<g5g, 32, 0, stream>>>(wh, FhT, H, 1024, 512, S_PT);
    k_stats<<<1024, 256, 0, stream>>>(H, st, 1024, S_PT);
    k_pool<<<BB*1024, 256, 0, stream>>>(H, st, g5, b5, z);

    k_fc<<<(BB*256 + 255)/256, 256, 0, stream>>>(z,  lw1, lb1, t1, 2048, 256);
    k_fc<<<(BB*64  + 255)/256, 256, 0, stream>>>(t1, lw2, lb2, t2, 256, 64);
    k_fc<<<(BB*10  + 255)/256, 256, 0, stream>>>(t2, lw3, lb3, o_e[which], 64, 10);

    k_cluster<<<BB, 256, 0, stream>>>(o_e[which], cw, o_q[which],
                                      which == 0 ? o_xd : (float*)nullptr);
  }
  k_sim<<<1, 32, 0, stream>>>(o_e[0], o_e[1], o_sim);

  (void)in_sizes; (void)n_in; (void)out_size; (void)ws_size;
}
